// SelfCluster_3083786519287
// MI455X (gfx1250) — matrix operand order repaired and checked by reading, not compiled, not run on hardware
//
#include <hip/hip_runtime.h>
#include <hip/hip_bf16.h>
#include <stdint.h>

typedef __attribute__((ext_vector_type(16))) _Float16 v16h;
typedef __attribute__((ext_vector_type(8)))  _Float16 v8h;
typedef __attribute__((ext_vector_type(8)))  float    v8f;
typedef int v4i_t __attribute__((vector_size(16)));   // matches builtin param pointee

#define N_     8
#define C_     256
#define H_     128
#define W_     128
#define HW_    (H_*W_)       // 16384
#define NP_    (N_*HW_)      // 131072 pixels
#define D_     512           // 2*c
#define M_     128           // n*fc*fh*fw
#define SC_    128           // 2c/fc
#define L_     4096          // sh*sw
#define S_     64            // AH*AW
#define HALF_  64

// ---------------- CDNA5 async global->LDS (guarded) ----------------
#if __has_builtin(__builtin_amdgcn_global_load_async_to_lds_b128)
#define HAS_ASYNC_LDS 1
#endif

__device__ __forceinline__ void async_copy16(void* dst_lds, const void* src_g) {
#ifdef HAS_ASYNC_LDS
  __builtin_amdgcn_global_load_async_to_lds_b128(
      (__attribute__((address_space(1))) v4i_t*)(unsigned long long)(uintptr_t)src_g,
      (__attribute__((address_space(3))) v4i_t*)(unsigned int)(uintptr_t)dst_lds,
      0, 0);
#else
  *(float4*)dst_lds = *(const float4*)src_g;   // sync fallback: load + ds_store
#endif
}
template<int NUM>
__device__ __forceinline__ void async_join() {
#if __has_builtin(__builtin_amdgcn_s_wait_asynccnt)
  __builtin_amdgcn_s_wait_asynccnt(NUM);
#endif
}

__device__ __forceinline__ float fast_rcp(float x) {
#if __has_builtin(__builtin_amdgcn_rcpf)
  return __builtin_amdgcn_rcpf(x);      // v_rcp_f32
#else
  return 1.0f / x;
#endif
}

// ---------------- WMMA fragment loaders ----------------
// A 16x32 f16 layout: lane holds row=lane&15; halfs K = (lane>>4)*8+{0..7} and 16+(lane>>4)*8+{0..7}
__device__ __forceinline__ v16h frag_a_lds(const unsigned char* base, int row, int lane, int pitch) {
  const unsigned char* p = base + row * pitch;
  const int hs = (lane >> 4) * 16;                    // bytes
  union { v16h v; v8h h[2]; } f;
  f.h[0] = *(const v8h*)(p + hs);
  f.h[1] = *(const v8h*)(p + 32 + hs);
  return f.v;
}
__device__ __forceinline__ v16h frag_a_glob(const _Float16* rowp, int kb, int lane) {
  const int hs = (lane >> 4) * 8;                     // halfs
  union { v16h v; v8h h[2]; } f;
  f.h[0] = *(const v8h*)(rowp + kb + hs);
  f.h[1] = *(const v8h*)(rowp + kb + 16 + hs);
  return f.v;
}
// B 32x16 f16 layout: lane holds col=lane&15; halfs K = (lane>>4)*8+{0..7} and 16+(lane>>4)*8+{0..7}
__device__ __forceinline__ v16h frag_b_lds(const unsigned char* base, int col, int lane, int pitch, int kbytes) {
  const unsigned char* p = base + col * pitch + kbytes + (lane >> 4) * 16;
  union { v16h v; v8h h[2]; } f;
  f.h[0] = *(const v8h*)(p);
  f.h[1] = *(const v8h*)(p + 32);
  return f.v;
}

// ---------------- epilogue scatter helpers ----------------
__device__ __forceinline__ void store_mode0(float* __restrict__ out, int p, int d, float v) {
  // y[n,d,h,w] -> xt[mi][li][sc] (m=128, l=4096, sc=128)
  const int n_i = p >> 14, hw = p & 16383;
  const int h = hw >> 7, wv = hw & 127;
  const int fc = d >> 7, sc = d & 127;
  const int mi = (((n_i << 2) | fc) << 2) | (((h >> 6) & 1) << 1) | ((wv >> 6) & 1);
  const int li = ((h & 63) << 6) | (wv & 63);
  out[(((size_t)mi * L_ + li) << 7) + sc] = v;
}
__device__ __forceinline__ void store_mode1(float* __restrict__ out, int p, int d, float v) {
  const int n_i = p >> 14, hw = p & 16383;
  out[((size_t)n_i * C_ + d) * HW_ + hw] = v;
}

// ---------------- GEMM: C[p,d] = A[p,:]·B[d,:] + bias[d] ----------------
// 128(M) x 64(N) block tile, 256 threads (8 waves), each wave a 32x32 sub-tile
// (4 accumulators -> 4 WMMA per 8 ds_loads), K step 32, double-buffered async LDS.
template<int MODE>
__global__ __launch_bounds__(256) void gemm128x64(
    const _Float16* __restrict__ A,    // [Mrows][K] halfs, row-major
    const _Float16* __restrict__ B,    // [Ncols][K] halfs, row-major (i.e. W[d,c])
    const float*    __restrict__ bias, // [Ncols]
    float*          __restrict__ out,
    int K)
{
  __shared__ __align__(16) unsigned char sA[2][128 * 80];
  __shared__ __align__(16) unsigned char sB[2][64 * 80];
  const int t    = threadIdx.x;
  const int lane = t & 31;
  const int w    = t >> 5;
  const int m0   = blockIdx.x * 128;
  const int n0   = blockIdx.y * 64;

  // staging: A = 512 chunks of 16B (2 per thread), B = 256 chunks (1 per thread)
  const int ra0 = t >> 2;               // A rows for chunks t and t+256
  const int ra1 = ra0 + 64;
  const int qa  = t & 3;
  const size_t arow0 = (size_t)(m0 + ra0) * K;
  const size_t arow1 = (size_t)(m0 + ra1) * K;
  const size_t brow  = (size_t)(n0 + ra0) * K;

  auto stage = [&](int buf, int kb) {
    async_copy16(&sA[buf][ra0 * 80 + qa * 16], A + arow0 + kb + qa * 8);
    async_copy16(&sA[buf][ra1 * 80 + qa * 16], A + arow1 + kb + qa * 8);
    async_copy16(&sB[buf][ra0 * 80 + qa * 16], B + brow  + kb + qa * 8);
  };

  v8f c00 = {}, c01 = {}, c10 = {}, c11 = {};
  const int mrow0 = (w >> 1) * 32;      // 0,32,64,96
  const int ncol0 = (w & 1) * 32;       // 0,32
  const int ksteps = K >> 5;

  stage(0, 0);
  for (int ks = 0; ks < ksteps; ++ks) {
    if (ks + 1 < ksteps) {
      stage((ks + 1) & 1, (ks + 1) * 32);
      async_join<3>();                  // stage ks done; ks+1 (3 ops/wave) in flight
    } else {
      async_join<0>();
    }
    __syncthreads();
    const unsigned char* a = sA[ks & 1];
    const unsigned char* b = sB[ks & 1];
    v16h a0 = frag_a_lds(a, mrow0 + (lane & 15), lane, 80);
    v16h a1 = frag_a_lds(a, mrow0 + 16 + (lane & 15), lane, 80);
    v16h b0 = frag_b_lds(b, ncol0 + (lane & 15), lane, 80, 0);
    v16h b1 = frag_b_lds(b, ncol0 + 16 + (lane & 15), lane, 80, 0);
    c00 = __builtin_amdgcn_wmma_f32_16x16x32_f16(false, a0, false, b0, (short)0, c00, false, false);
    c01 = __builtin_amdgcn_wmma_f32_16x16x32_f16(false, a0, false, b1, (short)0, c01, false, false);
    c10 = __builtin_amdgcn_wmma_f32_16x16x32_f16(false, a1, false, b0, (short)0, c10, false, false);
    c11 = __builtin_amdgcn_wmma_f32_16x16x32_f16(false, a1, false, b1, (short)0, c11, false, false);
    __syncthreads();
  }

  const int colg0 = n0 + ncol0 + (lane & 15);
  const int colg1 = colg0 + 16;
  const float bv0 = bias[colg0];
  const float bv1 = bias[colg1];
  const int halfsel = (lane >> 4) * 8;
  #pragma unroll
  for (int rr = 0; rr < 8; ++rr) {
    const int p0 = m0 + mrow0 + rr + halfsel;
    const int p1 = p0 + 16;
    if (MODE == 0) {
      store_mode0(out, p0, colg0, c00[rr] + bv0);
      store_mode0(out, p0, colg1, c01[rr] + bv1);
      store_mode0(out, p1, colg0, c10[rr] + bv0);
      store_mode0(out, p1, colg1, c11[rr] + bv1);
    } else {
      store_mode1(out, p0, colg0, c00[rr] + bv0);
      store_mode1(out, p0, colg1, c01[rr] + bv1);
      store_mode1(out, p1, colg0, c10[rr] + bv0);
      store_mode1(out, p1, colg1, c11[rr] + bv1);
    }
  }
}

// ---------------- x (n,c,hw) f32 -> xh (p, c) f16, LDS 32x32 transpose ----------------
__global__ __launch_bounds__(256) void cvt_x_kernel(const float* __restrict__ x, _Float16* __restrict__ xh) {
  __shared__ float tile[32][33];
  const int t = threadIdx.x;
  const int p0 = blockIdx.x * 32;        // 32 pixels (never crosses n: 16384 % 32 == 0)
  const int c0 = blockIdx.y * 32;
  const int n_i = p0 >> 14;
  const int hw0 = p0 & 16383;
  const int col = t & 31;
  const int row4 = t >> 5;               // 0..7
  #pragma unroll
  for (int i = 0; i < 4; ++i) {
    const int c = row4 + i * 8;
    tile[c][col] = x[((size_t)n_i * C_ + (c0 + c)) * HW_ + hw0 + col];
  }
  __syncthreads();
  #pragma unroll
  for (int i = 0; i < 4; ++i) {
    const int pr = row4 + i * 8;
    xh[((size_t)(p0 + pr)) * C_ + c0 + col] = (_Float16)tile[col][pr];
  }
}

__global__ void cvt_f16_kernel(const float* __restrict__ src, _Float16* __restrict__ dst, int nel) {
  const int i = blockIdx.x * 256 + threadIdx.x;
  if (i < nel) dst[i] = (_Float16)src[i];
}

// ---------------- anchors: 8x8 mean pool over xt -> at[mi][si][c] ----------------
__global__ __launch_bounds__(256) void anchor_kernel(const float* __restrict__ xt, float* __restrict__ at) {
  const int i = blockIdx.x * 256 + threadIdx.x;   // M_*S_*SC_
  const int cch = i & 127;
  const int si  = (i >> 7) & 63;
  const int mi  = i >> 13;
  const int sh0 = (si >> 3) * 8, sw0 = (si & 7) * 8;
  const float* base = xt + ((size_t)mi * L_) * SC_ + cch;
  float acc = 0.f;
  #pragma unroll
  for (int rr = 0; rr < 8; ++rr)
    for (int qq = 0; qq < 8; ++qq)
      acc += base[(size_t)(((sh0 + rr) << 6) | (sw0 + qq)) * SC_];
  at[i] = acc * (1.f / 64.f);
}

// ---------------- L2-normalize channels [0,64) of 128-stride rows -> f16 ----------------
__global__ __launch_bounds__(256) void l2norm_kernel(const float* __restrict__ src,
                                                     _Float16* __restrict__ dst, int nrows) {
  const int wid  = (blockIdx.x * 256 + threadIdx.x) >> 5;
  const int lane = threadIdx.x & 31;
  if (wid >= nrows) return;
  const float* rp = src + (size_t)wid * 128;
  const float a = rp[lane], b = rp[lane + 32];
  float ss = a * a + b * b;
  #pragma unroll
  for (int off = 16; off; off >>= 1) ss += __shfl_xor(ss, off, 32);
  const float inv = 1.f / fmaxf(sqrtf(ss), 1e-12f);
  _Float16* dp = dst + (size_t)wid * 64;
  dp[lane]      = (_Float16)(a * inv);
  dp[lane + 32] = (_Float16)(b * inv);
}

// ---------------- sims = sigmoid(a*xpn·apn^T+b); rowwise max+argmax ----------------
__global__ __launch_bounds__(256) void sims_kernel(
    const _Float16* __restrict__ xpn,   // [M_*L_][64]
    const _Float16* __restrict__ apn,   // [M_][S_][64]
    const float* __restrict__ alpha_p, const float* __restrict__ beta_p,
    float* __restrict__ vals, int* __restrict__ idxseg)
{
  __shared__ __align__(16) unsigned char sB[64 * 144];
  const int t = threadIdx.x;
  const int lane = t & 31;
  const int w = t >> 5;
  const int mi = blockIdx.y;
  const int rbase = blockIdx.x * 128;

  { // stage apn[mi] (64x64 halfs) into LDS, pitch 144B
    const _Float16* src = apn + (size_t)mi * (S_ * 64);
    #pragma unroll
    for (int i = 0; i < 2; ++i) {
      const int ch = t + i * 256;           // 0..511 chunks of 16B
      const int col = ch >> 3, qo = ch & 7;
      async_copy16(&sB[col * 144 + qo * 16], src + col * 64 + qo * 8);
    }
  }
  async_join<0>();
  __syncthreads();

  const float alpha = *alpha_p, beta = *beta_p;
  const int wrow = rbase + w * 16;
  v8f acc[4] = {};
  const _Float16* arow = xpn + ((size_t)mi * L_ + wrow + (lane & 15)) * 64;
  #pragma unroll
  for (int kb = 0; kb < 64; kb += 32) {
    const v16h af = frag_a_glob(arow, kb, lane);
    #pragma unroll
    for (int j = 0; j < 4; ++j) {
      const v16h bf = frag_b_lds(sB, j * 16 + (lane & 15), lane, 144, kb * 2);
      acc[j] = __builtin_amdgcn_wmma_f32_16x16x32_f16(false, af, false, bf, (short)0, acc[j], false, false);
    }
  }
  #pragma unroll
  for (int rr = 0; rr < 8; ++rr) {
    float bv = -1e30f; int bc = 0;
    #pragma unroll
    for (int j = 0; j < 4; ++j) {
      float s = acc[j][rr];
      s = fast_rcp(1.f + __expf(-(alpha * s + beta)));   // sigmoid via v_rcp_f32
      const int col = j * 16 + (lane & 15);
      if (s > bv || (s == bv && col < bc)) { bv = s; bc = col; }
    }
    #pragma unroll
    for (int off = 1; off < 16; off <<= 1) {
      const float ov = __shfl_xor(bv, off, 32);
      const int   oc = __shfl_xor(bc, off, 32);
      if (ov > bv || (ov == bv && oc < bc)) { bv = ov; bc = oc; }
    }
    if ((lane & 15) == 0) {
      const int rowg = mi * L_ + wrow + rr + ((lane >> 4) << 3);
      vals[rowg] = bv;
      idxseg[rowg] = mi * S_ + bc;
    }
  }
}

// ---------------- agg init / scatter / normalize / gather ----------------
__global__ void agg_init_kernel(const float* __restrict__ at, float* __restrict__ agg) {
  const int i = blockIdx.x * 256 + threadIdx.x;     // M_*S_*65
  if (i >= M_ * S_ * 65) return;
  const int g = i / 65, k = i % 65;
  agg[i] = (k == 64) ? 1.0f : at[(size_t)g * 128 + 64 + k];
}

__global__ __launch_bounds__(256) void scatter_kernel(const float* __restrict__ xt,
    const float* __restrict__ vals, const int* __restrict__ idxseg, float* __restrict__ agg) {
  const int wid  = (blockIdx.x * 256 + threadIdx.x) >> 5;  // row over M_*L_
  const int lane = threadIdx.x & 31;
  const float v = vals[wid];
  const int seg = idxseg[wid];
  const float* xv = xt + (size_t)wid * 128 + 64;
  float* ab = agg + (size_t)seg * 65;
  atomicAdd(&ab[lane],      v * xv[lane]);
  atomicAdd(&ab[lane + 32], v * xv[lane + 32]);
  if (lane == 0) atomicAdd(&ab[64], v);
}

__global__ void aggnorm_kernel(const float* __restrict__ agg, float* __restrict__ aggn) {
  const int i = blockIdx.x * 256 + threadIdx.x;     // M_*S_*64
  const int g = i >> 6, k = i & 63;
  aggn[i] = agg[(size_t)g * 65 + k] / agg[(size_t)g * 65 + 64];
}

__global__ __launch_bounds__(256) void disp_kernel(const float* __restrict__ vals,
    const int* __restrict__ idxseg, const float* __restrict__ aggn, _Float16* __restrict__ dh) {
  const size_t i = (size_t)blockIdx.x * 256 + threadIdx.x;  // NP_*256
  const int ch = (int)(i & 255);
  const size_t p = i >> 8;
  const int n_i = (int)(p >> 14);
  const int hw = (int)(p & 16383);
  const int h = hw >> 7, wv = hw & 127;
  const int fc = ch >> 6, k = ch & 63;
  const int mi = (((n_i << 2) | fc) << 2) | (((h >> 6) & 1) << 1) | ((wv >> 6) & 1);
  const int li = ((h & 63) << 6) | (wv & 63);
  const size_t row = (size_t)mi * L_ + li;
  dh[i] = (_Float16)(vals[row] * aggn[(size_t)idxseg[row] * 64 + k]);
}

// ---------------- host launcher ----------------
extern "C" void kernel_launch(void* const* d_in, const int* in_sizes, int n_in,
                              void* d_out, int out_size, void* d_ws, size_t ws_size,
                              hipStream_t stream) {
  const float* x       = (const float*)d_in[0];
  const float* proj_w  = (const float*)d_in[1];
  const float* proj_b  = (const float*)d_in[2];
  const float* merge_w = (const float*)d_in[3];
  const float* merge_b = (const float*)d_in[4];
  const float* alpha   = (const float*)d_in[5];
  const float* beta    = (const float*)d_in[6];
  float* out = (float*)d_out;

  char* ws = (char*)d_ws;
  size_t off = 0;
  auto carve = [&](size_t bytes) -> char* {
    char* p = ws + off;
    off += (bytes + 255) & ~(size_t)255;
    return p;
  };
  float*    xt   = (float*)   carve((size_t)M_ * L_ * SC_ * 4);   // 256 MiB
  _Float16* xh   = (_Float16*)carve((size_t)NP_ * C_ * 2);        // 64 MiB (reused as disp^T f16)
  float*    at   = (float*)   carve((size_t)M_ * S_ * SC_ * 4);
  _Float16* xpn  = (_Float16*)carve((size_t)M_ * L_ * HALF_ * 2);
  _Float16* apn  = (_Float16*)carve((size_t)M_ * S_ * HALF_ * 2);
  float*    vals = (float*)   carve((size_t)M_ * L_ * 4);
  int*      idxs = (int*)     carve((size_t)M_ * L_ * 4);
  float*    agg  = (float*)   carve((size_t)M_ * S_ * 65 * 4);
  float*    aggn = (float*)   carve((size_t)M_ * S_ * 64 * 4);
  _Float16* pwh  = (_Float16*)carve((size_t)D_ * C_ * 2);
  _Float16* mwh  = (_Float16*)carve((size_t)C_ * C_ * 2);

  cvt_x_kernel<<<dim3(NP_ / 32, C_ / 32), 256, 0, stream>>>(x, xh);
  cvt_f16_kernel<<<(D_ * C_ + 255) / 256, 256, 0, stream>>>(proj_w, pwh, D_ * C_);
  cvt_f16_kernel<<<(C_ * C_ + 255) / 256, 256, 0, stream>>>(merge_w, mwh, C_ * C_);

  gemm128x64<0><<<dim3(NP_ / 128, D_ / 64), 256, 0, stream>>>(xh, pwh, proj_b, xt, C_);

  anchor_kernel<<<(M_ * S_ * SC_) / 256, 256, 0, stream>>>(xt, at);
  l2norm_kernel<<<(M_ * L_) / 8, 256, 0, stream>>>(xt, xpn, M_ * L_);
  l2norm_kernel<<<(M_ * S_) / 8, 256, 0, stream>>>(at, apn, M_ * S_);
  agg_init_kernel<<<(M_ * S_ * 65 + 255) / 256, 256, 0, stream>>>(at, agg);

  sims_kernel<<<dim3(L_ / 128, M_), 256, 0, stream>>>(xpn, apn, alpha, beta, vals, idxs);
  scatter_kernel<<<(M_ * L_) / 8, 256, 0, stream>>>(xt, vals, idxs, agg);
  aggnorm_kernel<<<(M_ * S_ * 64) / 256, 256, 0, stream>>>(agg, aggn);
  disp_kernel<<<NP_, 256, 0, stream>>>(vals, idxs, aggn, xh);

  gemm128x64<1><<<dim3(NP_ / 128, C_ / 64), 256, 0, stream>>>(xh, mwh, merge_b, out, C_);
}
